// SemanticCompressor_80066780332247
// MI455X (gfx1250) — matrix operand order repaired and checked by reading, not compiled, not run on hardware
//
#include <hip/hip_runtime.h>
#include <hip/hip_bf16.h>
#include <math.h>

typedef unsigned int u32;
typedef unsigned short u16;
typedef __attribute__((ext_vector_type(16))) __bf16 v16bf;
typedef __attribute__((ext_vector_type(8)))  float   v8f;
typedef __attribute__((ext_vector_type(4)))  u32     u32x4;

#define HID   4096
#define HALFD 2048
#define BATCH 8
#define SEQ   1024
#define ROWS  (BATCH * SEQ)        // 8192
#define THRESH 0.85f

union Frag { u32x4 q[2]; v16bf v; };

__device__ __forceinline__ u16 f2bf(float f) {
  u32 u = __float_as_uint(f);
  u += 0x7FFFu + ((u >> 16) & 1u);   // round-to-nearest-even
  return (u16)(u >> 16);
}

__device__ __forceinline__ float gelu_exact(float x) {
  return 0.5f * x * (1.0f + erff(x * 0.70710678118654752f));
}

__device__ __forceinline__ Frag load_a(const u16* __restrict__ aptr, int k) {
  Frag f;
  f.q[0] = *(const u32x4*)(aptr + k);        // K+0..7   (or 8..15)
  f.q[1] = *(const u32x4*)(aptr + k + 16);   // K+16..23 (or 24..31)
  return f;
}

__device__ __forceinline__ void load_b4(const u16* __restrict__ bptr, size_t bstride,
                                        int k, Frag fb[4]) {
#pragma unroll
  for (int t = 0; t < 4; ++t) {
    const u16* bp = bptr + t * bstride;
    fb[t].q[0] = *(const u32x4*)(bp + k);      // K+0..7   (or 8..15)
    fb[t].q[1] = *(const u32x4*)(bp + k + 16); // K+16..23 (or 24..31)
  }
}

__device__ __forceinline__ void wmma4(const Frag& fa, const Frag fb[4], v8f acc[4]) {
#pragma unroll
  for (int t = 0; t < 4; ++t) {
    acc[t] = __builtin_amdgcn_wmma_f32_16x16x32_bf16(false, fa.v, false, fb[t].v,
                                                     (short)0, acc[t], false, false);
  }
}

// ---- one wave computes a 16x64 fp32 strip (4 WMMA tiles along N) of
// ---- A[M,K] (row-major bf16) x Bt[N,K]^T (Bt row-major bf16 = B transposed).
// ---- Two-stage software pipeline: loads for stage X are issued a full
// ---- half-iteration before the WMMAs that consume them.
__device__ __forceinline__ void wmma_strip4(const u16* __restrict__ A,
                                            const u16* __restrict__ Bt,
                                            int K, int m0, int n0, v8f acc[4]) {
  int lane = threadIdx.x & 31;
  int hi   = lane >> 4;          // 0: lanes 0-15, 1: lanes 16-31
  int lm   = lane & 15;
  const u16* aptr = A  + (size_t)(m0 + lm) * K + hi * 8;   // A frag K offsets
  const u16* bptr = Bt + (size_t)(n0 + lm) * K + hi * 8;   // B frag K offsets
  const size_t bstride = (size_t)16 * K;                   // next N sub-tile
#pragma unroll
  for (int t = 0; t < 4; ++t)
    acc[t] = (v8f){0.f, 0.f, 0.f, 0.f, 0.f, 0.f, 0.f, 0.f};

  // prologue: stage 0 for k = 0
  Frag fa0 = load_a(aptr, 0);
  Frag fb0[4];
  load_b4(bptr, bstride, 0, fb0);

  Frag fa1;
  Frag fb1[4];
  int k = 0;
#pragma unroll 1
  for (; k < K - 64; k += 64) {
    // issue stage-1 loads (k+32), then consume stage 0
    fa1 = load_a(aptr, k + 32);
    load_b4(bptr, bstride, k + 32, fb1);
    wmma4(fa0, fb0, acc);
    // issue stage-0 loads (k+64), then consume stage 1
    fa0 = load_a(aptr, k + 64);
    load_b4(bptr, bstride, k + 64, fb0);
    wmma4(fa1, fb1, acc);
  }
  // epilogue: k == K-64
  fa1 = load_a(aptr, k + 32);
  load_b4(bptr, bstride, k + 32, fb1);
  wmma4(fa0, fb0, acc);
  wmma4(fa1, fb1, acc);
}

// -------------------- conversion kernels --------------------
// W[R,C] fp32 row-major -> out[C,R] bf16 row-major (transpose + convert)
__global__ void k_transpose_cvt(const float* __restrict__ W, u16* __restrict__ out,
                                int R, int C) {
  size_t idx = (size_t)blockIdx.x * 256 + threadIdx.x;
  size_t n = (size_t)R * C;
  if (idx >= n) return;
  int r = (int)(idx / C), c = (int)(idx % C);
  out[(size_t)c * R + r] = f2bf(W[idx]);
}

__global__ void k_cvt_seg(const float* __restrict__ seg, u16* __restrict__ segb,
                          size_t n) {
  size_t idx = (size_t)blockIdx.x * 256 + threadIdx.x;
  if (idx >= n) return;
  segb[idx] = f2bf(seg[idx]);
}

__global__ void k_zero_u32(u32* __restrict__ p, size_t n) {
  size_t idx = (size_t)blockIdx.x * 256 + threadIdx.x;
  if (idx < n) p[idx] = 0u;
}

// -------------------- GEMM 1: p = gelu(seg @ W_sim + b_sim) --------------------
// block covers 16 rows x 512 cols (8 waves x 16x64 strips)
__global__ void k_gemm_p(const u16* __restrict__ A, const u16* __restrict__ Bt,
                         const float* __restrict__ bias, float* __restrict__ P) {
  const int nb_n = HALFD / 512;                       // 4
  int mb = blockIdx.x / nb_n;
  int nb = blockIdx.x % nb_n;
  int m0 = mb * 16;
  int n0 = nb * 512 + (threadIdx.x >> 5) * 64;
  v8f acc[4];
  wmma_strip4(A, Bt, HID, m0, n0, acc);
  int lane = threadIdx.x & 31;
  int hi   = (lane >> 4) * 8;
#pragma unroll
  for (int t = 0; t < 4; ++t) {
    int col = n0 + t * 16 + (lane & 15);
    float bv = bias[col];
#pragma unroll
    for (int r = 0; r < 8; ++r) {
      int row = m0 + r + hi;
      P[(size_t)row * HALFD + col] = gelu_exact(acc[t][r] + bv);
    }
  }
}

// -------------------- LayerNorm + L2 normalize -> bf16 --------------------
__device__ __forceinline__ float block_reduce(float v, float* red) {
  int t = threadIdx.x;
  red[t] = v;
  __syncthreads();
  for (int off = 128; off > 0; off >>= 1) {
    if (t < off) red[t] += red[t + off];
    __syncthreads();
  }
  float r = red[0];
  __syncthreads();
  return r;
}

__global__ void k_ln_norm(const float* __restrict__ P, u16* __restrict__ out,
                          const float* __restrict__ g, const float* __restrict__ b) {
  __shared__ float red[256];
  int row = blockIdx.x;
  const float* x = P + (size_t)row * HALFD;
  int t = threadIdx.x;
  float v[8];
  float s = 0.f, s2 = 0.f;
#pragma unroll
  for (int k = 0; k < 8; ++k) {
    float f = x[t + k * 256];
    v[k] = f; s += f; s2 += f * f;
  }
  float S  = block_reduce(s, red);
  float S2 = block_reduce(s2, red);
  float mean = S * (1.0f / HALFD);
  float var  = S2 * (1.0f / HALFD) - mean * mean;
  float rs   = rsqrtf(var + 1e-5f);
  float l2 = 0.f;
#pragma unroll
  for (int k = 0; k < 8; ++k) {
    int c = t + k * 256;
    float y = (v[k] - mean) * rs * g[c] + b[c];
    v[k] = y;
    l2 += y * y;
  }
  float L2 = block_reduce(l2, red);
  float inv = 1.0f / (sqrtf(L2) + 1e-8f);
#pragma unroll
  for (int k = 0; k < 8; ++k) {
    int c = t + k * 256;
    out[(size_t)row * HALFD + c] = f2bf(v[k] * inv);
  }
}

// -------------------- sim GEMM -> condition bitmask --------------------
// cond[b,i,j] bit set iff sim[b,i,j] > THRESH && imp[b,j] >= imp[b,i]
__global__ void k_gemm_sim(const u16* __restrict__ norm, const float* __restrict__ imp,
                           u32* __restrict__ cond) {
  const int nb_n = SEQ / 512;                         // 2
  const int bpb  = (SEQ / 16) * nb_n;                 // 128 blocks per batch
  int bidx = blockIdx.x / bpb;
  int rem  = blockIdx.x % bpb;
  int mb = rem / nb_n;
  int nb = rem % nb_n;
  int m0 = mb * 16;
  int n0 = nb * 512 + (threadIdx.x >> 5) * 64;
  const u16* base = norm + (size_t)bidx * SEQ * HALFD;
  v8f acc[4];
  wmma_strip4(base, base, HALFD, m0, n0, acc);
  int lane = threadIdx.x & 31;
  int hi   = (lane >> 4) * 8;
#pragma unroll
  for (int t = 0; t < 4; ++t) {
    int j = n0 + t * 16 + (lane & 15);
    float impj = imp[(size_t)bidx * SEQ + j];
#pragma unroll
    for (int r = 0; r < 8; ++r) {
      int i = m0 + r + hi;
      float impi = imp[(size_t)bidx * SEQ + i];
      if (acc[t][r] > THRESH && impj >= impi) {
        atomicOr(&cond[((size_t)bidx * SEQ + i) * (SEQ / 32) + (j >> 5)],
                 1u << (j & 31));
      }
    }
  }
}

// -------------------- sequential greedy suppression (one wave32 / batch) ----
__global__ void k_greedy(const u32* __restrict__ cond, float* __restrict__ kept) {
  int b = blockIdx.x;
  int lane = threadIdx.x;                 // 0..31, one word of 32 j's each
  u32 kw = 0xFFFFFFFFu;                   // kept bits for j = lane*32 .. +31
  const u32* cb = cond + (size_t)b * SEQ * (SEQ / 32);
  for (int i = 1; i < SEQ; ++i) {
    u32 cword = cb[(size_t)i * (SEQ / 32) + lane];
    int lo = lane * 32;
    u32 vmask;
    if (i >= lo + 32)      vmask = 0xFFFFFFFFu;
    else if (i <= lo)      vmask = 0u;
    else                   vmask = (1u << (i - lo)) - 1u;
    u32 hit = cword & kw & vmask;
    if (__any(hit != 0u)) {
      if (lane == (i >> 5)) kw &= ~(1u << (i & 31));
    }
  }
  float* out = kept + (size_t)b * SEQ;
#pragma unroll
  for (int t = 0; t < 32; ++t) {
    out[lane * 32 + t] = ((kw >> t) & 1u) ? 1.0f : 0.0f;
  }
}

// -------------------- GEMM d1: h = gelu(imp*(seg @ W_d1) + b_d1) -> bf16 ----
__global__ void k_gemm_d1(const u16* __restrict__ A, const u16* __restrict__ Bt,
                          const float* __restrict__ imp, const float* __restrict__ bias,
                          u16* __restrict__ H) {
  const int nb_n = HALFD / 512;                       // 4
  int mb = blockIdx.x / nb_n;
  int nb = blockIdx.x % nb_n;
  int m0 = mb * 16;
  int n0 = nb * 512 + (threadIdx.x >> 5) * 64;
  v8f acc[4];
  wmma_strip4(A, Bt, HID, m0, n0, acc);
  int lane = threadIdx.x & 31;
  int hi   = (lane >> 4) * 8;
#pragma unroll
  for (int t = 0; t < 4; ++t) {
    int col = n0 + t * 16 + (lane & 15);
    float bv = bias[col];
#pragma unroll
    for (int r = 0; r < 8; ++r) {
      int row = m0 + r + hi;
      float x = acc[t][r] * imp[row] + bv;   // imp*(seg@W) == (imp*seg)@W
      H[(size_t)row * HALFD + col] = f2bf(gelu_exact(x));
    }
  }
}

// -------------------- GEMM d2: out = (h @ W_d2 + b_d2) * kept --------------
__global__ void k_gemm_d2(const u16* __restrict__ A, const u16* __restrict__ Bt,
                          const float* __restrict__ bias, const float* __restrict__ kept,
                          float* __restrict__ O) {
  const int nb_n = HID / 512;                         // 8
  int mb = blockIdx.x / nb_n;
  int nb = blockIdx.x % nb_n;
  int m0 = mb * 16;
  int n0 = nb * 512 + (threadIdx.x >> 5) * 64;
  v8f acc[4];
  wmma_strip4(A, Bt, HALFD, m0, n0, acc);
  int lane = threadIdx.x & 31;
  int hi   = (lane >> 4) * 8;
#pragma unroll
  for (int t = 0; t < 4; ++t) {
    int col = n0 + t * 16 + (lane & 15);
    float bv = bias[col];
#pragma unroll
    for (int r = 0; r < 8; ++r) {
      int row = m0 + r + hi;
      O[(size_t)row * HID + col] = (acc[t][r] + bv) * kept[row];
    }
  }
}

// ============================ launch ============================
extern "C" void kernel_launch(void* const* d_in, const int* in_sizes, int n_in,
                              void* d_out, int out_size, void* d_ws, size_t ws_size,
                              hipStream_t stream) {
  const float* seg   = (const float*)d_in[0];   // [8,1024,4096]
  const float* imp   = (const float*)d_in[1];   // [8,1024]
  const float* W_sim = (const float*)d_in[2];   // [4096,2048]
  const float* b_sim = (const float*)d_in[3];   // [2048]
  const float* ln_g  = (const float*)d_in[4];   // [2048]
  const float* ln_b  = (const float*)d_in[5];   // [2048]
  const float* W_d1  = (const float*)d_in[6];   // [4096,2048]
  const float* b_d1  = (const float*)d_in[7];   // [2048]
  const float* W_d2  = (const float*)d_in[8];   // [2048,4096]
  const float* b_d2  = (const float*)d_in[9];   // [4096]

  float* out_comp = (float*)d_out;                              // [8192,4096]
  float* out_kept = (float*)d_out + (size_t)ROWS * HID;         // [8192]

  char* ws = (char*)d_ws;
  const size_t MB = 1024ull * 1024ull;
  u16* wsimT = (u16*)(ws + 0 * MB);     // [2048,4096] bf16  (16 MB)
  u16* wd1T  = (u16*)(ws + 16 * MB);    // [2048,4096] bf16  (16 MB)
  u16* wd2T  = (u16*)(ws + 32 * MB);    // [4096,2048] bf16  (16 MB)
  u16* segb  = (u16*)(ws + 48 * MB);    // [8192,4096] bf16  (64 MB)
  float* P   = (float*)(ws + 112 * MB); // [8192,2048] f32   (64 MB)
  u16* H     = (u16*)(ws + 112 * MB);   // [8192,2048] bf16  (reuses P, 32 MB)
  u32* cond  = (u32*)(ws + 144 * MB);   // [8192,32] bits    (reuses P tail, 1 MB)
  u16* normb = (u16*)(ws + 176 * MB);   // [8192,2048] bf16  (32 MB)

  // 1) weight transpose+convert to bf16
  {
    size_t n = (size_t)HID * HALFD;
    int blocks = (int)((n + 255) / 256);
    k_transpose_cvt<<<blocks, 256, 0, stream>>>(W_sim, wsimT, HID, HALFD);
    k_transpose_cvt<<<blocks, 256, 0, stream>>>(W_d1,  wd1T,  HID, HALFD);
    k_transpose_cvt<<<blocks, 256, 0, stream>>>(W_d2,  wd2T,  HALFD, HID);
  }
  // 2) seg -> bf16
  {
    size_t n = (size_t)ROWS * HID;
    k_cvt_seg<<<(int)((n + 255) / 256), 256, 0, stream>>>(seg, segb, n);
  }
  // 3) p = gelu(seg @ W_sim + b_sim)   [8192x2048, K=4096]
  k_gemm_p<<<(ROWS / 16) * (HALFD / 512), 256, 0, stream>>>(segb, wsimT, b_sim, P);
  // 4) LayerNorm + L2 normalize -> bf16 norm
  k_ln_norm<<<ROWS, 256, 0, stream>>>(P, normb, ln_g, ln_b);
  // 5) zero cond bitmask, then sim GEMM -> cond bits
  k_zero_u32<<<(int)(((size_t)ROWS * 32 + 255) / 256), 256, 0, stream>>>(cond, (size_t)ROWS * 32);
  k_gemm_sim<<<BATCH * (SEQ / 16) * (SEQ / 512), 256, 0, stream>>>(normb, imp, cond);
  // 6) sequential greedy suppression -> kept mask (written straight to d_out tail)
  k_greedy<<<BATCH, 32, 0, stream>>>(cond, out_kept);
  // 7) h = gelu(imp*(seg @ W_d1) + b_d1) -> bf16   [8192x2048, K=4096]
  k_gemm_d1<<<(ROWS / 16) * (HALFD / 512), 256, 0, stream>>>(segb, wd1T, imp, b_d1, H);
  // 8) out = (h @ W_d2 + b_d2) * kept   [8192x4096, K=2048]
  k_gemm_d2<<<(ROWS / 16) * (HID / 512), 256, 0, stream>>>(H, wd2T, b_d2, out_kept, out_comp);
}
